// ForecastPFN_PAttn_3453153706385
// MI455X (gfx1250) — matrix operand order repaired and checked by reading, not compiled, not run on hardware
//
#include <hip/hip_runtime.h>

typedef __attribute__((ext_vector_type(16))) _Float16 v16h;
typedef __attribute__((ext_vector_type(8)))  _Float16 v8h;
typedef __attribute__((ext_vector_type(8)))  float    v8f;

#define Bsz   256
#define Tlen  100
#define SEQ   101
#define DM    72
#define NH    4
#define EH    18
#define PN    12
#define DFF   1152
#define TOK   221184      /* Bsz*DM*PN */
#define MT2   6912        /* TOK/32: two 16-row tiles per wave */
#define KPAD  96          /* DM padded for WMMA K */
#define NPAD  80          /* DM padded for WMMA N */

// ---------------- helpers ----------------

__device__ __forceinline__ float blockReduceSum(float v, float* buf) {
    int t = threadIdx.x;
    buf[t] = v; __syncthreads();
    for (int s = 64; s > 0; s >>= 1) { if (t < s) buf[t] += buf[t + s]; __syncthreads(); }
    float r = buf[0]; __syncthreads();
    return r;
}

// positional encoding channel c (0..35) given the four time indices
__device__ __forceinline__ float pe_val(int c, int iy, int im, int idd, int iw) {
    float P; int i, j; bool is_cos;
    if (c < 8)       { P = 10.f; i = iy;  j = (c < 4)  ? c      : c - 4;  is_cos = (c >= 4);  }
    else if (c < 16) { P = 12.f; i = im;  j = (c < 12) ? c - 8  : c - 12; is_cos = (c >= 12); }
    else if (c < 28) { P = 31.f; i = idd; j = (c < 22) ? c - 16 : c - 22; is_cos = (c >= 22); }
    else             { P = 7.f;  i = iw;  j = (c < 32) ? c - 28 : c - 32; is_cos = (c >= 32); }
    float ang = (3.14159265358979323846f / P) * exp2f((float)j) * ((float)i - 1.0f);
    return is_cos ? cosf(ang) : sinf(ang);
}

// build a v16h A-operand fragment from two 16-byte chunks (ISA A layout)
__device__ __forceinline__ v16h mk16(const _Float16* p0, const _Float16* p1) {
    v8h lo = *(const v8h*)p0;
    v8h hi = *(const v8h*)p1;
    v16h r;
#pragma unroll
    for (int e = 0; e < 8; ++e) { r[e] = lo[e]; r[e + 8] = hi[e]; }
    return r;
}

// ---------------- K0: weight prep (f16, transposed+padded for WMMA B operands) ----------------
// w1t[n][k] : n in [0,1152), k in [0,96)  (K pad 72->96)
// w2t[n][k] : n in [0,80),   k in [0,1152) (N pad 72->80)
__global__ void k_prep_w(const float* __restrict__ w_ff1, const float* __restrict__ w_ff2,
                         _Float16* __restrict__ w1t, _Float16* __restrict__ w2t) {
    int i = blockIdx.x * 256 + threadIdx.x;
    if (i < DFF * KPAD) {
        int n = i / KPAD, k = i % KPAD;
        w1t[i] = (_Float16)((k < DM) ? w_ff1[k * DFF + n] : 0.0f);
    } else {
        int j = i - DFF * KPAD;
        if (j < NPAD * DFF) {
            int n = j / DFF, k = j % DFF;
            w2t[j] = (_Float16)((n < DM) ? w_ff2[k * DM + n] : 0.0f);
        }
    }
}

// ---------------- K1: masked stats -> scale, embeddings -> hist(B,101,72) ----------------
__global__ __launch_bounds__(128)
void k_pre(const float* __restrict__ hx,
           const float* __restrict__ w_nopos, const float* __restrict__ b_nopos,
           const float* __restrict__ w_wpos,  const float* __restrict__ b_wpos,
           const float* __restrict__ task_emb,
           const int* __restrict__ ts, const int* __restrict__ target_ts,
           const int* __restrict__ task,
           float* __restrict__ hist, float* __restrict__ outp) {
    __shared__ float sx[Tlen];
    __shared__ float red[128];
    int b = blockIdx.x, t = threadIdx.x;
    const float* x = hx + b * Tlen;
    if (t < Tlen) sx[t] = x[t];
    __syncthreads();

    // pass 1: masked mean/std of x (mask: x != 0)
    float ps = 0.f, pc = 0.f;
    for (int i = t; i < Tlen; i += 128) { float v = sx[i]; if (v != 0.f) { ps += v; pc += 1.f; } }
    float sum = blockReduceSum(ps, red);
    float cnt = blockReduceSum(pc, red);
    float safe = fmaxf(cnt, 1.f);
    float mean = sum / safe;
    float pv = 0.f;
    for (int i = t; i < Tlen; i += 128) { float v = sx[i]; if (v != 0.f) { float d = v - mean; pv += d * d; } }
    float var = blockReduceSum(pv, red) / safe;
    float stdv = sqrtf(var);
    if (cnt <= 0.f) { mean = 0.f; stdv = 0.f; }
    float hiclip = mean + 2.f * stdv;

    // pass 2: masked mean/std of clipped
    ps = 0.f; pc = 0.f;
    for (int i = t; i < Tlen; i += 128) {
        float cv = fminf(fmaxf(sx[i], 0.f), hiclip);
        if (cv != 0.f) { ps += cv; pc += 1.f; }
    }
    sum = blockReduceSum(ps, red);
    cnt = blockReduceSum(pc, red);
    safe = fmaxf(cnt, 1.f);
    float mc = sum / safe;
    pv = 0.f;
    for (int i = t; i < Tlen; i += 128) {
        float cv = fminf(fmaxf(sx[i], 0.f), hiclip);
        if (cv != 0.f) { float d = cv - mc; pv += d * d; }
    }
    float sc = sqrtf(blockReduceSum(pv, red) / safe);
    if (cnt <= 0.f) { mc = 0.f; sc = 0.f; }
    float s = mc + sc + 1e-4f;
    if (t == 0) outp[Bsz + b] = s;       // scale output (second return value)
    float inv_s = 1.f / s;

    int yr_last = ts[b * 400 + 99 * 4 + 0];

    // rows 0..99
    for (int o = t; o < Tlen * 36; o += 128) {
        int tt = o / 36, c = o % 36;
        float h = fminf(fmaxf(sx[tt] * inv_s, 0.f), 3.f);
        float e0 = fmaxf(h * w_nopos[c] + b_nopos[c], 0.f);
        const int* tp = ts + b * 400 + tt * 4;
        int iy = min(max(yr_last - tp[0], 0), 10);
        float pos = pe_val(c, iy, tp[1], tp[2], tp[3]);
        float e1 = fmaxf(h * w_wpos[c] + b_wpos[c], 0.f) + pos;
        float* hr = hist + ((long)b * SEQ + tt) * DM;
        hr[c] = e0; hr[36 + c] = e1;
    }
    // target row (t = 100)
    if (t < 36) {
        int c = t;
        float tv = task_emb[task[b] * 36 + c];
        const int* tt = target_ts + b * 5;
        int iy = min(max(yr_last - tt[0], 0), 10);
        float qp = pe_val(c, iy, tt[1], tt[2], tt[3]);
        float* hr = hist + ((long)b * SEQ + Tlen) * DM;
        hr[c] = tv; hr[36 + c] = tv + qp;
    }
}

// ---------------- K2: time-LN + patching + patch GEMM -> z(TOK,72) ----------------
__global__ __launch_bounds__(256)
void k_patch(const float* __restrict__ hist, const float* __restrict__ w_inp,
             const float* __restrict__ b_inp, float* __restrict__ z) {
    __shared__ float sh[SEQ * DM];
    __shared__ float sm[DM], si[DM];
    __shared__ float sw[16 * DM + DM];
    int b = blockIdx.x, t = threadIdx.x;
    for (int i = t; i < SEQ * DM; i += 256) sh[i] = hist[(long)b * SEQ * DM + i];
    for (int i = t; i < 16 * DM; i += 256) sw[i] = w_inp[i];
    if (t < DM) sw[16 * DM + t] = b_inp[t];
    __syncthreads();
    if (t < DM) {
        float s = 0.f;
        for (int r = 0; r < SEQ; ++r) s += sh[r * DM + t];
        float m = s * (1.f / SEQ), v = 0.f;
        for (int r = 0; r < SEQ; ++r) { float d = sh[r * DM + t] - m; v += d * d; }
        sm[t] = m; si[t] = rsqrtf(v * (1.f / SEQ) + 1e-5f);
    }
    __syncthreads();
    for (int i = t; i < SEQ * DM; i += 256) { int c = i % DM; sh[i] = (sh[i] - sm[c]) * si[c]; }
    __syncthreads();
    for (int o = t; o < DM * PN * DM; o += 256) {
        int c = o / (PN * DM), rem = o % (PN * DM), p = rem / DM, d = rem % DM;
        float acc = sw[16 * DM + d];
#pragma unroll
        for (int i2 = 0; i2 < 16; ++i2) {
            int pos = p * 8 + i2; if (pos > 100) pos = 100;   // xp repeats last column
            acc += sh[pos * DM + c] * sw[i2 * DM + d];
        }
        z[((long)(b * DM + c) * PN + p) * DM + d] = acc;
    }
}

// ---------------- K3: fused QKV + attention + O + residual + LN1 -> z1 / z1h ----------------
__global__ __launch_bounds__(128)
void k_attn(const float* __restrict__ z,
            const float* __restrict__ wq, const float* __restrict__ bq,
            const float* __restrict__ wk, const float* __restrict__ bk,
            const float* __restrict__ wv, const float* __restrict__ bv,
            const float* __restrict__ wo, const float* __restrict__ bo,
            const float* __restrict__ g1, const float* __restrict__ be1,
            float* __restrict__ z1, _Float16* __restrict__ z1h) {
    __shared__ float zb[PN * DM], qs[PN * DM], ks_[PN * DM], vs[PN * DM], os_[PN * DM];
    __shared__ float rm[PN], ri[PN];
    int bd = blockIdx.x, t = threadIdx.x;
    const float* zp = z + (long)bd * PN * DM;
    for (int i = t; i < PN * DM; i += 128) zb[i] = zp[i];
    __syncthreads();
    // QKV projections
    for (int i = t; i < 3 * PN * DM; i += 128) {
        int w = i / (PN * DM), rem = i % (PN * DM), p = rem / DM, d = rem % DM;
        const float* W = (w == 0) ? wq : ((w == 1) ? wk : wv);
        const float* Bb = (w == 0) ? bq : ((w == 1) ? bk : bv);
        float acc = Bb[d];
        for (int kk = 0; kk < DM; ++kk) acc += zb[p * DM + kk] * W[kk * DM + d];
        ((w == 0) ? qs : ((w == 1) ? ks_ : vs))[rem] = acc;
    }
    __syncthreads();
    // attention: one (head, query) per thread
    if (t < NH * PN) {
        int h = t / PN, l = t % PN;
        float sc[PN]; float mx = -1e30f;
#pragma unroll
        for (int s2 = 0; s2 < PN; ++s2) {
            float a = 0.f;
#pragma unroll
            for (int e = 0; e < EH; ++e) a += qs[l * DM + h * EH + e] * ks_[s2 * DM + h * EH + e];
            a *= 0.23570226039551584f;   // 1/sqrt(18)
            sc[s2] = a; mx = fmaxf(mx, a);
        }
        float den = 0.f;
#pragma unroll
        for (int s2 = 0; s2 < PN; ++s2) { sc[s2] = expf(sc[s2] - mx); den += sc[s2]; }
        float idn = 1.f / den;
#pragma unroll
        for (int e = 0; e < EH; ++e) {
            float a = 0.f;
#pragma unroll
            for (int s2 = 0; s2 < PN; ++s2) a += sc[s2] * vs[s2 * DM + h * EH + e];
            os_[l * DM + h * EH + e] = a * idn;
        }
    }
    __syncthreads();
    // O projection + residual (into qs, reused)
    for (int i = t; i < PN * DM; i += 128) {
        int p = i / DM, d = i % DM;
        float acc = bo[d] + zb[i];
        for (int kk = 0; kk < DM; ++kk) acc += os_[p * DM + kk] * wo[kk * DM + d];
        qs[i] = acc;
    }
    __syncthreads();
    if (t < PN) {
        float s = 0.f;
        for (int d = 0; d < DM; ++d) s += qs[t * DM + d];
        float m = s * (1.f / DM), v = 0.f;
        for (int d = 0; d < DM; ++d) { float df = qs[t * DM + d] - m; v += df * df; }
        rm[t] = m; ri[t] = rsqrtf(v * (1.f / DM) + 1e-5f);
    }
    __syncthreads();
    for (int i = t; i < PN * KPAD; i += 128) {
        int p = i / KPAD, d = i % KPAD;
        long tok = (long)bd * PN + p;
        float val = 0.f;
        if (d < DM) {
            val = (qs[p * DM + d] - rm[p]) * ri[p] * g1[d] + be1[d];
            z1[tok * DM + d] = val;
        }
        z1h[tok * KPAD + d] = (_Float16)val;
    }
}

// ---------------- K4: fused FF1 -> gelu -> FF2 (WMMA), 2 M-tiles per wave ----------------
// Each B-operand fragment is loaded once and feeds two WMMAs (one per M-tile),
// halving L2 weight traffic and interleaving independent WMMA chains to cover
// the WMMA->WMMA RAW hazard latency.
__global__ __launch_bounds__(32)
void k_ffn(const _Float16* __restrict__ z1h, const float* __restrict__ z1,
           const _Float16* __restrict__ w1t, const _Float16* __restrict__ w2t,
           const float* __restrict__ b_ff1, const float* __restrict__ b_ff2,
           const float* __restrict__ g2, const float* __restrict__ be2,
           const float* __restrict__ ge, const float* __restrict__ bee,
           float* __restrict__ z2) {
    __shared__ __align__(16) _Float16 sG[2 * 16 * 32];   // gelu blocks, A-layout staging
    __shared__ float sOut[2 * 16 * NPAD];
    const int tid = threadIdx.x;
    const int lane = tid & 15, half = tid >> 4;
    const long m0 = (long)blockIdx.x * 32;

    // A fragments for both 16-row tiles: K = 96 (3 k-steps of 32)
    v16h a0[3], a1[3];
    {
        const _Float16* zr0 = z1h + (m0 + lane) * KPAD;
        const _Float16* zr1 = z1h + (m0 + 16 + lane) * KPAD;
#pragma unroll
        for (int ks = 0; ks < 3; ++ks) {
            a0[ks] = mk16(zr0 + ks * 32 + half * 8, zr0 + ks * 32 + 16 + half * 8);
            a1[ks] = mk16(zr1 + ks * 32 + half * 8, zr1 + ks * 32 + 16 + half * 8);
        }
    }

    v8f acc0[5], acc1[5];
#pragma unroll
    for (int nt = 0; nt < 5; ++nt)
#pragma unroll
        for (int r = 0; r < 8; ++r) { acc0[nt][r] = 0.f; acc1[nt][r] = 0.f; }

    for (int ch = 0; ch < 36; ++ch) {
        if (ch + 1 < 36) {  // hint next chunk of FF1 weights toward the caches
            __builtin_prefetch(w1t + (size_t)((ch + 1) * 32 + half * 16 + lane) * KPAD, 0, 3);
        }
        // GEMM1: 16x32 chunk of FF1 output for both tiles, K = 96
        v8f c1a[2], c1b[2];
#pragma unroll
        for (int jn = 0; jn < 2; ++jn) {
#pragma unroll
            for (int r = 0; r < 8; ++r) { c1a[jn][r] = 0.f; c1b[jn][r] = 0.f; }
            const _Float16* wp = w1t + (size_t)(ch * 32 + jn * 16 + lane) * KPAD + half * 8;
#pragma unroll
            for (int ks = 0; ks < 3; ++ks) {
                v16h bf = mk16(wp + ks * 32, wp + ks * 32 + 16);
                c1a[jn] = __builtin_amdgcn_wmma_f32_16x16x32_f16(
                    false, a0[ks], false, bf, (short)0, c1a[jn], false, false);
                c1b[jn] = __builtin_amdgcn_wmma_f32_16x16x32_f16(
                    false, a1[ks], false, bf, (short)0, c1b[jn], false, false);
            }
        }
        // bias + exact gelu -> LDS (transpose to A layout), both tiles
#pragma unroll
        for (int jn = 0; jn < 2; ++jn) {
            float bias = b_ff1[ch * 32 + jn * 16 + lane];
#pragma unroll
            for (int r = 0; r < 8; ++r) {
                float xa = c1a[jn][r] + bias;
                float xb = c1b[jn][r] + bias;
                float ga = 0.5f * xa * (1.0f + erff(xa * 0.70710678118f));
                float gb = 0.5f * xb * (1.0f + erff(xb * 0.70710678118f));
                int rr = (r + 8 * half) * 32 + jn * 16 + lane;
                sG[rr] = (_Float16)ga;
                sG[512 + rr] = (_Float16)gb;
            }
        }
        __syncthreads();
        v16h a2_0 = mk16(&sG[lane * 32 + half * 8], &sG[lane * 32 + 16 + half * 8]);
        v16h a2_1 = mk16(&sG[512 + lane * 32 + half * 8], &sG[512 + lane * 32 + 16 + half * 8]);
        __syncthreads();
        // GEMM2: accumulate into two 16x80 C tiles, K = 32 (exact)
#pragma unroll
        for (int nt = 0; nt < 5; ++nt) {
            v16h b2 = mk16(w2t + (size_t)(nt * 16 + lane) * DFF + ch * 32 + half * 8, w2t + (size_t)(nt * 16 + lane) * DFF + ch * 32 + 16 + half * 8);
            acc0[nt] = __builtin_amdgcn_wmma_f32_16x16x32_f16(
                false, a2_0, false, b2, (short)0, acc0[nt], false, false);
            acc1[nt] = __builtin_amdgcn_wmma_f32_16x16x32_f16(
                false, a2_1, false, b2, (short)0, acc1[nt], false, false);
        }
    }

    // dump accumulators to LDS (C layout -> row-major), both tiles
#pragma unroll
    for (int nt = 0; nt < 5; ++nt)
#pragma unroll
        for (int r = 0; r < 8; ++r) {
            sOut[(r + 8 * half) * NPAD + nt * 16 + lane] = acc0[nt][r];
            sOut[(16 + r + 8 * half) * NPAD + nt * 16 + lane] = acc1[nt][r];
        }
    __syncthreads();

    // epilogue: one row per lane: + b_ff2 + residual(z1), LN(g2,be2), LN(ge,bee)
    {
        long token = m0 + tid;
        float* row = sOut + tid * NPAD;
        const float* zr1 = z1 + token * DM;
        float s1 = 0.f;
        for (int d = 0; d < DM; ++d) { float v = row[d] + b_ff2[d] + zr1[d]; row[d] = v; s1 += v; }
        float mean = s1 * (1.f / DM), s2 = 0.f;
        for (int d = 0; d < DM; ++d) { float df = row[d] - mean; s2 += df * df; }
        float inv = rsqrtf(s2 * (1.f / DM) + 1e-5f);
        for (int d = 0; d < DM; ++d) row[d] = (row[d] - mean) * inv * g2[d] + be2[d];
        s1 = 0.f;
        for (int d = 0; d < DM; ++d) s1 += row[d];
        mean = s1 * (1.f / DM); s2 = 0.f;
        for (int d = 0; d < DM; ++d) { float df = row[d] - mean; s2 += df * df; }
        inv = rsqrtf(s2 * (1.f / DM) + 1e-5f);
        float* zo = z2 + token * DM;
        for (int d = 0; d < DM; ++d) zo[d] = (row[d] - mean) * inv * ge[d] + bee[d];
    }
}

// ---------------- K5: output projection + head ----------------
__global__ __launch_bounds__(128)
void k_head(const float* __restrict__ z2, const float* __restrict__ w_outp,
            const float* __restrict__ b_outp, const float* __restrict__ w_head,
            const float* __restrict__ b_head, float* __restrict__ outp) {
    __shared__ float so[DM];
    int b = blockIdx.x, t = threadIdx.x;
    if (t < DM) {
        const float* zp = z2 + (long)b * DM * PN * DM + (long)t * PN * DM;
        float acc = b_outp[0];
        for (int j = 0; j < PN * DM; ++j) acc += zp[j] * w_outp[j];
        so[t] = acc;
    }
    __syncthreads();
    if (t == 0) {
        float y = b_head[0];
        for (int c = 0; c < DM; ++c) y += so[c] * w_head[c];
        y = fmaxf(y, 0.f);
        outp[b] = y * outp[Bsz + b];   // * scale
    }
}

// ---------------- launch ----------------
extern "C" void kernel_launch(void* const* d_in, const int* in_sizes, int n_in,
                              void* d_out, int out_size, void* d_ws, size_t ws_size,
                              hipStream_t stream) {
    (void)in_sizes; (void)n_in; (void)out_size; (void)ws_size;
    const float* history  = (const float*)d_in[0];
    const float* w_nopos  = (const float*)d_in[1];
    const float* b_nopos  = (const float*)d_in[2];
    const float* w_wpos   = (const float*)d_in[3];
    const float* b_wpos   = (const float*)d_in[4];
    const float* task_emb = (const float*)d_in[5];
    const float* w_inp    = (const float*)d_in[6];
    const float* b_inp    = (const float*)d_in[7];
    const float* wq = (const float*)d_in[8];   const float* bq = (const float*)d_in[9];
    const float* wk = (const float*)d_in[10];  const float* bk = (const float*)d_in[11];
    const float* wv = (const float*)d_in[12];  const float* bv = (const float*)d_in[13];
    const float* wo = (const float*)d_in[14];  const float* bo = (const float*)d_in[15];
    const float* w_ff1 = (const float*)d_in[16]; const float* b_ff1 = (const float*)d_in[17];
    const float* w_ff2 = (const float*)d_in[18]; const float* b_ff2 = (const float*)d_in[19];
    const float* g1  = (const float*)d_in[20]; const float* be1 = (const float*)d_in[21];
    const float* g2  = (const float*)d_in[22]; const float* be2 = (const float*)d_in[23];
    const float* ge  = (const float*)d_in[24]; const float* bee = (const float*)d_in[25];
    const float* w_outp = (const float*)d_in[26]; const float* b_outp = (const float*)d_in[27];
    const float* w_head = (const float*)d_in[28]; const float* b_head = (const float*)d_in[29];
    const int* ts        = (const int*)d_in[30];
    const int* target_ts = (const int*)d_in[31];
    const int* task      = (const int*)d_in[32];
    float* outp = (float*)d_out;

    // workspace layout (bytes)
    char* ws = (char*)d_ws;
    const size_t off_hist = 0;                                        // 7,446,528
    const size_t off_z    = off_hist + (size_t)Bsz * SEQ * DM * 4;    // 63,700,992
    const size_t off_z1   = off_z    + (size_t)TOK * DM * 4;          // 63,700,992
    const size_t off_z1h  = off_z1   + (size_t)TOK * DM * 4;          // 42,467,328
    const size_t off_w1t  = off_z1h  + (size_t)TOK * KPAD * 2;        //    221,184
    const size_t off_w2t  = off_w1t  + (size_t)DFF * KPAD * 2;        //    184,320
    float*    hist = (float*)(ws + off_hist);
    float*    z    = (float*)(ws + off_z);
    float*    z1   = (float*)(ws + off_z1);
    _Float16* z1h  = (_Float16*)(ws + off_z1h);
    _Float16* w1t  = (_Float16*)(ws + off_w1t);
    _Float16* w2t  = (_Float16*)(ws + off_w2t);
    float*    z2   = z;   // z is dead after K3; reuse for z2

    k_prep_w<<<(DFF * KPAD + NPAD * DFF + 255) / 256, 256, 0, stream>>>(w_ff1, w_ff2, w1t, w2t);
    k_pre<<<Bsz, 128, 0, stream>>>(history, w_nopos, b_nopos, w_wpos, b_wpos,
                                   task_emb, ts, target_ts, task, hist, outp);
    k_patch<<<Bsz, 256, 0, stream>>>(hist, w_inp, b_inp, z);
    k_attn<<<Bsz * DM, 128, 0, stream>>>(z, wq, bq, wk, bk, wv, bv, wo, bo, g1, be1, z1, z1h);
    k_ffn<<<MT2, 32, 0, stream>>>(z1h, z1, w1t, w2t, b_ff1, b_ff2, g2, be2, ge, bee, z2);
    k_head<<<Bsz, 128, 0, stream>>>(z2, w_outp, b_outp, w_head, b_head, outp);
}
